// neural_network_equivariant_1425929142514
// MI455X (gfx1250) — matrix operand order repaired and checked by reading, not compiled, not run on hardware
//
#include <hip/hip_runtime.h>
#include <cstdint>

#define NSC 64
#define NVC 32
#define NBB 8
#define NLAY 4
#define EMBD 32
#define NWW 192
#define MAX_RF 3.0f
#define PI_F 3.14159265358979f
#define INV_SQRT_NEIGH 0.17677669529663687f
#define WAVES 4

typedef __bf16 v16bf __attribute__((ext_vector_type(16)));
typedef __bf16 v2bf  __attribute__((ext_vector_type(2)));
typedef float  v8f   __attribute__((ext_vector_type(8)));

__device__ __forceinline__ v8f wmma_bf16(v16bf a, v16bf b, v8f c) {
  return __builtin_amdgcn_wmma_f32_16x16x32_bf16(false, a, false, b, (short)0, c, false, false);
}

// A-operand (16x32 bf16) per-lane element -> K index within 32-chunk (contiguous order k = kh*16 + idx, matched by the B packing)
__device__ __forceinline__ int kidxA(int idx, int kh) {
  return kh * 16 + idx;
}

// fast-math helpers (v_rcp_f32 / v_exp_f32 / v_sin_f32 / v_cos_f32)
__device__ __forceinline__ float rcp_fast(float x) { return __builtin_amdgcn_rcpf(x); }
__device__ __forceinline__ float sigmoid_fast(float x) { return rcp_fast(1.f + __expf(-x)); }
__device__ __forceinline__ float silu_fast(float x) { return x * sigmoid_fast(x); }

__device__ __forceinline__ float smooth_cutoff_f(float xx) {
  float u = 2.f * (xx - 1.f);
  float y = (1.f - __cosf(PI_F * u)) * 0.5f;
  y = (u > 0.f) ? 0.f : y;
  y = (u < -1.f) ? 1.f : y;
  return y;
}

// ---------------------------------------------------------------------------
// Per-layer weight prep: f32 -> bf16, PRE-SWIZZLED into WMMA B-fragment order.
// frag[(kb*ntl + nt)*32 + lane][16], element idx -> (K=kb*32+kh*16+idx, col=nt*16+n)
// ---------------------------------------------------------------------------
__global__ void convert_weights(const float* __restrict__ fc1, const float* __restrict__ fc2,
                                const float* __restrict__ W_sv, const float* __restrict__ W_vs,
                                const float* __restrict__ W_sc_s, const float* __restrict__ W_sc_v,
                                const float* __restrict__ Ws, const float* __restrict__ Wv,
                                __bf16* __restrict__ fc1b, __bf16* __restrict__ fc2b,
                                __bf16* __restrict__ wsv0b, __bf16* __restrict__ wsv1b,
                                __bf16* __restrict__ wvsb,
                                __bf16* __restrict__ wscsb, __bf16* __restrict__ wscvb,
                                __bf16* __restrict__ wsb, __bf16* __restrict__ wvb, int layer) {
  int tid = blockIdx.x * blockDim.x + threadIdx.x;
  auto coord = [](int t, int ntl, int& K, int& col) {
    int idx = t & 15, lane = (t >> 4) & 31, blk = t >> 9;
    int nt = blk % ntl, kb = blk / ntl;
    int kh = lane >> 4, n = lane & 15;
    K = kb * 32 + kh * 16 + idx;
    col = nt * 16 + n;
  };
  if (tid < 2048) {  // fc1: K=32 (padded from 16), N=64
    int K, col; coord(tid, 4, K, col);
    fc1b[tid] = (__bf16)(K < 16 ? fc1[layer * 16 * 64 + K * 64 + col] : 0.f);
    return;
  }
  tid -= 2048;
  if (tid < 12288) {  // fc2: K=64, N=192
    int K, col; coord(tid, 12, K, col);
    fc2b[tid] = (__bf16)fc2[layer * 64 * NWW + K * NWW + col];
    return;
  }
  tid -= 12288;
  if (tid < 2048) {  // W_sv[:,:,0,:]: K=64, N=32
    int K, col; coord(tid, 2, K, col);
    wsv0b[tid] = (__bf16)W_sv[((layer * NSC + K) * 2 + 0) * NVC + col];
    return;
  }
  tid -= 2048;
  if (tid < 2048) {  // W_sv[:,:,1,:]
    int K, col; coord(tid, 2, K, col);
    wsv1b[tid] = (__bf16)W_sv[((layer * NSC + K) * 2 + 1) * NVC + col];
    return;
  }
  tid -= 2048;
  if (tid < 6144) {  // W_vs flat: K=64, N=96
    int K, col; coord(tid, 6, K, col);
    wvsb[tid] = (__bf16)W_vs[layer * 64 * 96 + K * 96 + col];
    return;
  }
  tid -= 6144;
  if (tid < 196608) {  // W_sc_s flat: K=2048, N=96
    int K, col; coord(tid, 6, K, col);
    wscsb[tid] = (__bf16)W_sc_s[layer * 2048 * 96 + K * 96 + col];
    return;
  }
  tid -= 196608;
  if (tid < 32768) {  // W_sc_v flat: K=1024, N=32
    int K, col; coord(tid, 2, K, col);
    wscvb[tid] = (__bf16)W_sc_v[layer * 1024 * 32 + K * 32 + col];
    return;
  }
  tid -= 32768;
  if (tid < 4096) {  // Ws: K=64, N=64
    int K, col; coord(tid, 4, K, col);
    wsb[tid] = (__bf16)Ws[layer * 64 * 64 + K * 64 + col];
    return;
  }
  tid -= 4096;
  if (tid < 1024) {  // Wv: K=32, N=32
    int K, col; coord(tid, 2, K, col);
    wvb[tid] = (__bf16)Wv[layer * 32 * 32 + K * 32 + col];
    return;
  }
}

// ---------------------------------------------------------------------------
// Node-side A-matrix builders (swizzled A-fragment bf16 layout):
// Asc[n, u*32+e] = s[n,u]*emb[n,e]   (K=2048, Kc=64) ; Sa = s (K=64, Kc=2)
// Ascv[n*3+c, u*32+e] = v[n,u,c]*emb[n,e] (K=1024, Kc=32) ; Va (K=32, Kc=1)
// ---------------------------------------------------------------------------
__global__ void build_Asc(const float* __restrict__ s, const float* __restrict__ emb,
                          __bf16* __restrict__ Asc, __bf16* __restrict__ Sa, int N) {
  int tid = blockIdx.x * blockDim.x + threadIdx.x;
  int Mt = (N + 15) >> 4;
  int totalA = Mt * 64 * 512;
  if (tid < totalA) {
    int idx = tid & 15, lane = (tid >> 4) & 31, blk = tid >> 9;
    int kb = blk & 63, mt = blk >> 6;
    int row = mt * 16 + (lane & 15); if (row >= N) row = N - 1;
    int k = kb * 32 + kidxA(idx, lane >> 4);
    Asc[tid] = (__bf16)(s[row * NSC + (k >> 5)] * emb[row * EMBD + (k & 31)]);
    return;
  }
  tid -= totalA;
  if (tid < Mt * 2 * 512) {
    int idx = tid & 15, lane = (tid >> 4) & 31, blk = tid >> 9;
    int kb = blk & 1, mt = blk >> 1;
    int row = mt * 16 + (lane & 15); if (row >= N) row = N - 1;
    int k = kb * 32 + kidxA(idx, lane >> 4);
    Sa[tid] = (__bf16)s[row * NSC + k];
  }
}

__global__ void build_Ascv(const float* __restrict__ v, const float* __restrict__ emb,
                           __bf16* __restrict__ Ascv, __bf16* __restrict__ Va, int N) {
  int tid = blockIdx.x * blockDim.x + threadIdx.x;
  int M = N * 3;
  int Mt = (M + 15) >> 4;
  int totalA = Mt * 32 * 512;
  if (tid < totalA) {
    int idx = tid & 15, lane = (tid >> 4) & 31, blk = tid >> 9;
    int kb = blk & 31, mt = blk >> 5;
    int row = mt * 16 + (lane & 15); if (row >= M) row = M - 1;
    int n = row / 3, c = row % 3;
    int k = kb * 32 + kidxA(idx, lane >> 4);
    Ascv[tid] = (__bf16)(v[(n * NVC + (k >> 5)) * 3 + c] * emb[n * EMBD + (k & 31)]);
    return;
  }
  tid -= totalA;
  if (tid < Mt * 512) {
    int idx = tid & 15, lane = (tid >> 4) & 31, blk = tid >> 9;
    int mt = blk;
    int row = mt * 16 + (lane & 15); if (row >= M) row = M - 1;
    int n = row / 3, c = row % 3;
    int k = kidxA(idx, lane >> 4);
    Va[tid] = (__bf16)v[(n * NVC + k) * 3 + c];
  }
}

// ---------------------------------------------------------------------------
// Generic WMMA GEMM over pre-swizzled operands. One wave per 16-row M tile,
// kb-outer so A fragments load once per K-chunk; NTL accumulators in regs.
// ---------------------------------------------------------------------------
template <int NTL>
__global__ __launch_bounds__(128) void gemm_swz(const __bf16* __restrict__ A,
                                                const __bf16* __restrict__ B,
                                                float* __restrict__ C,
                                                int Mtiles, int Kc, int NC, int Mrows) {
  const int wid = threadIdx.x >> 5, lane = threadIdx.x & 31;
  const int m = lane & 15, kh = lane >> 4;
  const v16bf* Af = (const v16bf*)A;
  const v16bf* Bf = (const v16bf*)B;
  for (int mt = blockIdx.x * 4 + wid; mt < Mtiles; mt += gridDim.x * 4) {
    v8f zero = {};
    v8f acc[NTL];
#pragma unroll
    for (int nt = 0; nt < NTL; ++nt) acc[nt] = zero;
    for (int kb = 0; kb < Kc; ++kb) {
      v16bf a = Af[(mt * Kc + kb) * 32 + lane];
#pragma unroll
      for (int nt = 0; nt < NTL; ++nt)
        acc[nt] = wmma_bf16(a, Bf[(kb * NTL + nt) * 32 + lane], acc[nt]);
    }
#pragma unroll
    for (int nt = 0; nt < NTL; ++nt)
#pragma unroll
      for (int rr = 0; rr < 8; ++rr) {
        int row = mt * 16 + rr + kh * 8;
        if (row < Mrows) C[row * NC + nt * 16 + m] = acc[nt][rr];
      }
  }
}

// ---------------------------------------------------------------------------
__global__ void uplift_kernel(const float* __restrict__ x, const float* __restrict__ W_up,
                              float* __restrict__ v, int N) {
  int idx = blockIdx.x * blockDim.x + threadIdx.x;
  if (idx >= N * 96) return;
  int n = idx / 96, rem = idx % 96, vch = rem / 3, c = rem % 3;
  v[idx] = x[n * 6 + c] * W_up[vch] + x[n * 6 + 3 + c] * W_up[NVC + vch];
}

__global__ void gather_emb(const float* __restrict__ emb_table, const int* __restrict__ node_attr,
                           float* __restrict__ attr_emb, int N) {
  int idx = blockIdx.x * blockDim.x + threadIdx.x;
  if (idx >= N * EMBD) return;
  int n = idx / EMBD, e = idx % EMBD;
  attr_emb[idx] = emb_table[node_attr[n] * EMBD + e];
}

// ---------------------------------------------------------------------------
// Fused edge pipeline (unchanged structure; wstage now bf16 -> 82KB LDS/block)
// ---------------------------------------------------------------------------
__global__ __launch_bounds__(128) void edge_mega(
    const float* __restrict__ xcur, const int* __restrict__ esrc, const int* __restrict__ edst,
    const float* __restrict__ s, const float* __restrict__ v,
    const __bf16* __restrict__ fc1b, const float* __restrict__ b1p, const __bf16* __restrict__ fc2b,
    const __bf16* __restrict__ wsv0b, const __bf16* __restrict__ wsv1b, const __bf16* __restrict__ wvsb,
    float* __restrict__ agg_s, float* __restrict__ agg_v, int E) {
  __shared__ __align__(32) __bf16 sfc1[2048];
  __shared__ __align__(32) __bf16 sfc2[12288];
  __shared__ __align__(32) __bf16 ssv0[2048];
  __shared__ __align__(32) __bf16 ssv1[2048];
  __shared__ __align__(32) __bf16 svs[6144];
  __shared__ float sb1[64];
  __shared__ __align__(8) __bf16 wstage[WAVES][16 * NWW];  // 24 KB
  __shared__ __align__(32) __bf16 hstage[WAVES][1024];     //  8 KB
  __shared__ float  astage[WAVES][16 * 6];
  __shared__ int    dstage[WAVES][16];

  for (int i = threadIdx.x; i < 256; i += blockDim.x)  ((uint4*)sfc1)[i] = ((const uint4*)fc1b)[i];
  for (int i = threadIdx.x; i < 1536; i += blockDim.x) ((uint4*)sfc2)[i] = ((const uint4*)fc2b)[i];
  for (int i = threadIdx.x; i < 256; i += blockDim.x)  ((uint4*)ssv0)[i] = ((const uint4*)wsv0b)[i];
  for (int i = threadIdx.x; i < 256; i += blockDim.x)  ((uint4*)ssv1)[i] = ((const uint4*)wsv1b)[i];
  for (int i = threadIdx.x; i < 768; i += blockDim.x)  ((uint4*)svs)[i]  = ((const uint4*)wvsb)[i];
  if (threadIdx.x < 64) sb1[threadIdx.x] = b1p[threadIdx.x];
  __syncthreads();

  const v16bf* Ffc1 = (const v16bf*)sfc1;
  const v16bf* Ffc2 = (const v16bf*)sfc2;
  const v16bf* Fsv0 = (const v16bf*)ssv0;
  const v16bf* Fsv1 = (const v16bf*)ssv1;
  const v16bf* Fvs  = (const v16bf*)svs;

  const int wid = threadIdx.x >> 5;
  const int lane = threadIdx.x & 31;
  const int m = lane & 15;
  const int kh = lane >> 4;
  const int ntiles = (E + 15) >> 4;

  for (int tile = blockIdx.x * WAVES + wid; tile < ntiles; tile += gridDim.x * WAVES) {
    int e = tile * 16 + m;
    if (e >= E) e = E - 1;
    const int src = esrc[e];
    const int dst = edst[e];
    { int pe = (e + 16 < E) ? e + 16 : e; __builtin_prefetch(&esrc[pe], 0, 1); }

    float v0x = xcur[src * 6 + 0] - xcur[dst * 6 + 0];
    float v0y = xcur[src * 6 + 1] - xcur[dst * 6 + 1];
    float v0z = xcur[src * 6 + 2] - xcur[dst * 6 + 2];
    float v1x = xcur[src * 6 + 3] - xcur[dst * 6 + 3];
    float v1y = xcur[src * 6 + 4] - xcur[dst * 6 + 4];
    float v1z = xcur[src * 6 + 5] - xcur[dst * 6 + 5];
    float r0 = sqrtf(v0x * v0x + v0y * v0y + v0z * v0z);
    float r1 = sqrtf(v1x * v1x + v1y * v1y + v1z * v1z);
    float ir0 = rcp_fast(r0), ir1 = rcp_fast(r1);
    const float SQ3 = 1.7320508075688772f;
    float cf0 = smooth_cutoff_f(r0 * (1.f / MAX_RF)) * SQ3 * ir0;
    float cf1 = smooth_cutoff_f(r1 * (1.f / MAX_RF)) * SQ3 * ir1;
    float a0x = cf0 * v0x, a0y = cf0 * v0y, a0z = cf0 * v0z;
    float a1x = cf1 * v1x, a1y = cf1 * v1y, a1z = cf1 * v1z;
    if (kh == 0) {
      astage[wid][m * 6 + 0] = a0x; astage[wid][m * 6 + 1] = a0y; astage[wid][m * 6 + 2] = a0z;
      astage[wid][m * 6 + 3] = a1x; astage[wid][m * 6 + 4] = a1y; astage[wid][m * 6 + 5] = a1z;
      dstage[wid][m] = dst;
    }

    float w0 = (PI_F / MAX_RF) * r0, w1 = (PI_F / MAX_RF) * r1;
    v16bf afrag;
#pragma unroll
    for (int idx = 0; idx < 16; ++idx) {
      int K = kidxA(idx, kh);
      float fv = 0.f;
      if (K < 16) {
        float arg = (K < 8) ? w0 : w1;
        float ir = (K < 8) ? ir0 : ir1;
        float nb = (float)((K & 7) + 1);
        fv = 2.3094010767585033f * __sinf(nb * arg) * ir;
      }
      afrag[idx] = (__bf16)fv;
    }

    // GEMM1 -> hstage (A-frag layout)
#pragma unroll
    for (int nt = 0; nt < 4; ++nt) {
      v8f c = {};
      c = wmma_bf16(afrag, Ffc1[nt * 32 + lane], c);
#pragma unroll
      for (int rr = 0; rr < 8; ++rr) {
        int row = rr + kh * 8, col = nt * 16 + m;
        int kb2 = nt >> 1;
        int idx2 = m;
        int lane2 = (nt & 1) * 16 + row;
        hstage[wid][(kb2 * 32 + lane2) * 16 + idx2] = (__bf16)silu_fast(c[rr] + sb1[col]);
      }
    }

    // GEMM2 -> wstage (bf16)
    {
      v16bf a2_0 = ((const v16bf*)&hstage[wid][0])[lane];
      v16bf a2_1 = ((const v16bf*)&hstage[wid][0])[32 + lane];
#pragma unroll
      for (int nt = 0; nt < 12; ++nt) {
        v8f c = {};
        c = wmma_bf16(a2_0, Ffc2[nt * 32 + lane], c);
        c = wmma_bf16(a2_1, Ffc2[(12 + nt) * 32 + lane], c);
#pragma unroll
        for (int rr = 0; rr < 8; ++rr)
          wstage[wid][(rr + kh * 8) * NWW + nt * 16 + m] = (__bf16)c[rr];
      }
    }

    // scalar->vector: t fragments, q GEMMs, msg_v scatter
    {
      v16bf t0f[2], t1f[2];
#pragma unroll
      for (int kb = 0; kb < 2; ++kb) {
#pragma unroll
        for (int idx = 0; idx < 16; ++idx) {
          int k = kb * 32 + kidxA(idx, kh);
          float sv = s[src * NSC + k];
          v2bf wp = *(const v2bf*)&wstage[wid][m * NWW + 2 * k];
          t0f[kb][idx] = (__bf16)(sv * (float)wp[0]);
          t1f[kb][idx] = (__bf16)(sv * (float)wp[1]);
        }
      }
#pragma unroll
      for (int nt = 0; nt < 2; ++nt) {
        v8f c0 = {}, c1 = {};
#pragma unroll
        for (int kb = 0; kb < 2; ++kb) {
          c0 = wmma_bf16(t0f[kb], Fsv0[(kb * 2 + nt) * 32 + lane], c0);
          c1 = wmma_bf16(t1f[kb], Fsv1[(kb * 2 + nt) * 32 + lane], c1);
        }
#pragma unroll
        for (int rr = 0; rr < 8; ++rr) {
          int row = rr + kh * 8;
          int e2 = tile * 16 + row;
          if (e2 < E) {
            int d2 = dstage[wid][row];
            int mchan = nt * 16 + m;
#pragma unroll
            for (int c3 = 0; c3 < 3; ++c3) {
              float mv = (c0[rr] * astage[wid][row * 6 + c3] +
                          c1[rr] * astage[wid][row * 6 + 3 + c3]) * INV_SQRT_NEIGH;
              atomicAdd(&agg_v[(d2 * NVC + mchan) * 3 + c3], mv);
            }
          }
        }
      }
    }

    // vector->scalar: dw fragment, msg_s GEMM + scatter
    {
      v16bf dwf[2];
      const float* vbase = &v[src * NVC * 3];
#pragma unroll
      for (int kb = 0; kb < 2; ++kb) {
#pragma unroll
        for (int idx = 0; idx < 16; ++idx) {
          int k = kb * 32 + kidxA(idx, kh);
          int u = k >> 1, vv = k & 1;
          float ax = vv ? a1x : a0x, ay = vv ? a1y : a0y, az = vv ? a1z : a0z;
          float d = vbase[u * 3 + 0] * ax + vbase[u * 3 + 1] * ay + vbase[u * 3 + 2] * az;
          dwf[kb][idx] = (__bf16)(d * (float)wstage[wid][m * NWW + 128 + k]);
        }
      }
#pragma unroll
      for (int nt = 0; nt < 6; ++nt) {
        v8f c = {};
#pragma unroll
        for (int kb = 0; kb < 2; ++kb)
          c = wmma_bf16(dwf[kb], Fvs[(kb * 6 + nt) * 32 + lane], c);
#pragma unroll
        for (int rr = 0; rr < 8; ++rr) {
          int row = rr + kh * 8;
          int e2 = tile * 16 + row;
          if (e2 < E) {
            int d2 = dstage[wid][row];
            atomicAdd(&agg_s[d2 * 96 + nt * 16 + m], c[rr] * INV_SQRT_NEIGH);
          }
        }
      }
    }
  }
}

// ---------------------------------------------------------------------------
// Elementwise node update from precomputed GEMM outputs.
// ---------------------------------------------------------------------------
__global__ __launch_bounds__(128) void node_final(
    const float* __restrict__ s, const float* __restrict__ v,
    const float* __restrict__ s_prev, const float* __restrict__ v_prev,
    const float* __restrict__ agg_s, const float* __restrict__ agg_v,
    const float* __restrict__ SCs, const float* __restrict__ SCv,
    const float* __restrict__ SIs, const float* __restrict__ SIv,
    const float* __restrict__ hp, const float* __restrict__ mp,
    const float* __restrict__ W_proj,
    float* __restrict__ s_out, float* __restrict__ v_out, float* __restrict__ x_out, int N) {
  int n = blockIdx.x, t = threadIdx.x;
  __shared__ float gatel[NVC], vnl[NVC * 3];
  float hv = hp[0], mx = mp[0], h2 = hv * hv;
  if (t >= 64 && t < 96)
    gatel[t - 64] = sigmoid_fast(SCs[n * 96 + t] + agg_s[n * 96 + t]);
  __syncthreads();
  if (t < 64) {
    float cs = SCs[n * 96 + t] + agg_s[n * 96 + t];
    float gs = silu_fast(cs);
    s_out[n * 64 + t] = 2.f * s[n * 64 + t] - s_prev[n * 64 + t] +
                        h2 * (mx * gs + (mx - 1.f) * SIs[n * 64 + t]);
  }
  if (t < 96) {
    int c = t / 32, vch = t % 32;
    int oi = vch * 3 + c;
    int r = n * 3 + c;
    float cv = SCv[r * 32 + vch] + agg_v[n * 96 + oi];
    float gv = cv * gatel[vch];
    float vn = 2.f * v[n * 96 + oi] - v_prev[n * 96 + oi] +
               h2 * (mx * gv + (mx - 1.f) * SIv[r * 32 + vch]);
    v_out[n * 96 + oi] = vn;
    vnl[oi] = vn;
  }
  __syncthreads();
  if (t < 6) {
    int i2 = t / 3, cc = t % 3;
    float acc = 0.f;
    for (int u = 0; u < NVC; ++u) acc += vnl[u * 3 + cc] * W_proj[u * 2 + i2];
    x_out[n * 6 + i2 * 3 + cc] = acc;
  }
}

// ---------------------------------------------------------------------------
extern "C" void kernel_launch(void* const* d_in, const int* in_sizes, int n_in,
                              void* d_out, int out_size, void* d_ws, size_t ws_size,
                              hipStream_t stream) {
  const float* x         = (const float*)d_in[0];
  const float* emb_table = (const float*)d_in[1];
  const float* W_up      = (const float*)d_in[2];
  const float* W_proj    = (const float*)d_in[3];
  const float* fc1       = (const float*)d_in[4];
  const float* b1        = (const float*)d_in[5];
  const float* fc2       = (const float*)d_in[6];
  const float* W_sv      = (const float*)d_in[7];
  const float* W_vs      = (const float*)d_in[8];
  const float* W_sc_s    = (const float*)d_in[9];
  const float* W_sc_v    = (const float*)d_in[10];
  const float* Ws        = (const float*)d_in[11];
  const float* Wv        = (const float*)d_in[12];
  const float* hp        = (const float*)d_in[13];
  const float* mixp      = (const float*)d_in[14];
  const int* node_attr   = (const int*)d_in[15];
  const int* edge_src    = (const int*)d_in[16];
  const int* edge_dst    = (const int*)d_in[17];
  const int N = in_sizes[0] / 6;
  const int E = in_sizes[16];
  const int M3 = N * 3;
  const int Mt  = (N + 15) / 16;
  const int Mt3 = (M3 + 15) / 16;

  char* p = (char*)d_ws;
  auto alloc = [&](size_t bytes) -> void* {
    void* q = (void*)p;
    p += (bytes + 255) & ~(size_t)255;
    return q;
  };
  float* sA    = (float*)alloc((size_t)N * NSC * 4);
  float* sB    = (float*)alloc((size_t)N * NSC * 4);
  float* vA    = (float*)alloc((size_t)N * 96 * 4);
  float* vB    = (float*)alloc((size_t)N * 96 * 4);
  float* xA    = (float*)alloc((size_t)N * 6 * 4);
  float* xB    = (float*)alloc((size_t)N * 6 * 4);
  float* aemb  = (float*)alloc((size_t)N * EMBD * 4);
  float* aggS  = (float*)alloc((size_t)N * 96 * 4);
  float* aggV  = (float*)alloc((size_t)N * 96 * 4);
  __bf16* fc1b  = (__bf16*)alloc(2048 * 2);
  __bf16* fc2b  = (__bf16*)alloc(12288 * 2);
  __bf16* wsv0b = (__bf16*)alloc(2048 * 2);
  __bf16* wsv1b = (__bf16*)alloc(2048 * 2);
  __bf16* wvsb  = (__bf16*)alloc(6144 * 2);
  __bf16* wscsb = (__bf16*)alloc(196608 * 2);
  __bf16* wscvb = (__bf16*)alloc(32768 * 2);
  __bf16* wsb   = (__bf16*)alloc(4096 * 2);
  __bf16* wvb   = (__bf16*)alloc(1024 * 2);
  __bf16* Asc   = (__bf16*)alloc((size_t)Mt * 64 * 512 * 2);
  __bf16* Sa    = (__bf16*)alloc((size_t)Mt * 2 * 512 * 2);
  __bf16* Ascv  = (__bf16*)alloc((size_t)Mt3 * 32 * 512 * 2);
  __bf16* Va    = (__bf16*)alloc((size_t)Mt3 * 512 * 2);
  float* SCs   = (float*)alloc((size_t)N * 96 * 4);
  float* SCv   = (float*)alloc((size_t)M3 * 32 * 4);
  float* SIs   = (float*)alloc((size_t)N * 64 * 4);
  float* SIv   = (float*)alloc((size_t)M3 * 32 * 4);

  hipMemsetAsync(sA, 0, (size_t)N * NSC * 4, stream);
  hipMemsetAsync(sB, 0, (size_t)N * NSC * 4, stream);
  uplift_kernel<<<(N * 96 + 255) / 256, 256, 0, stream>>>(x, W_up, vA, N);
  hipMemcpyAsync(vB, vA, (size_t)N * 96 * 4, hipMemcpyDeviceToDevice, stream);
  gather_emb<<<(N * EMBD + 255) / 256, 256, 0, stream>>>(emb_table, node_attr, aemb, N);

  float* sCur = sA; float* sPrev = sB;
  float* vCur = vA; float* vPrev = vB;
  const float* xCur = x;
  const int ntiles = (E + 15) / 16;
  const int megaBlocks = (ntiles + WAVES - 1) / WAVES;
  const int cvtTotal = 2048 + 12288 + 2048 + 2048 + 6144 + 196608 + 32768 + 4096 + 1024;
  const int bAscTot  = Mt * 64 * 512 + Mt * 2 * 512;
  const int bAscvTot = Mt3 * 32 * 512 + Mt3 * 512;

  for (int i = 0; i < NLAY; ++i) {
    hipMemsetAsync(aggS, 0, (size_t)N * 96 * 4, stream);
    hipMemsetAsync(aggV, 0, (size_t)N * 96 * 4, stream);
    convert_weights<<<(cvtTotal + 255) / 256, 256, 0, stream>>>(
        fc1, fc2, W_sv, W_vs, W_sc_s, W_sc_v, Ws, Wv,
        fc1b, fc2b, wsv0b, wsv1b, wvsb, wscsb, wscvb, wsb, wvb, i);
    edge_mega<<<megaBlocks, 128, 0, stream>>>(
        xCur, edge_src, edge_dst, sCur, vCur,
        fc1b, b1 + (size_t)i * 64, fc2b, wsv0b, wsv1b, wvsb,
        aggS, aggV, E);
    // node-side GEMMs (WMMA)
    build_Asc<<<(bAscTot + 255) / 256, 256, 0, stream>>>(sCur, aemb, Asc, Sa, N);
    build_Ascv<<<(bAscvTot + 255) / 256, 256, 0, stream>>>(vCur, aemb, Ascv, Va, N);
    gemm_swz<6><<<(Mt + 3) / 4, 128, 0, stream>>>(Asc, wscsb, SCs, Mt, 64, 96, N);
    gemm_swz<2><<<(Mt3 + 3) / 4, 128, 0, stream>>>(Ascv, wscvb, SCv, Mt3, 32, 32, M3);
    gemm_swz<4><<<(Mt + 3) / 4, 128, 0, stream>>>(Sa, wsb, SIs, Mt, 2, 64, N);
    gemm_swz<2><<<(Mt3 + 3) / 4, 128, 0, stream>>>(Va, wvb, SIv, Mt3, 1, 32, M3);
    float* xOut = (i == NLAY - 1) ? (float*)d_out : ((i & 1) ? xB : xA);
    node_final<<<N, 128, 0, stream>>>(
        sCur, vCur, sPrev, vPrev, aggS, aggV, SCs, SCv, SIs, SIv,
        hp + i, mixp + i, W_proj,
        sPrev, vPrev, xOut, N);
    float* ts = sCur; sCur = sPrev; sPrev = ts;
    float* tv = vCur; vCur = vPrev; vPrev = tv;
    xCur = xOut;
  }
}
